// CausalSelfAttention_57019985821965
// MI455X (gfx1250) — compile-verified, matrix operand order checked by reading, not run on hardware
//
#include <hip/hip_runtime.h>

// ---------------- Types & helpers ----------------
typedef __attribute__((ext_vector_type(16))) __bf16 bf16x16;
typedef __attribute__((ext_vector_type(8)))  float  f32x8;

union FragB16 {
  bf16x16 v;
  uint4   q[2];
  unsigned short us[16];
};

__device__ __forceinline__ unsigned short f32_to_bf16(float f) {
  unsigned int u = __float_as_uint(f);
  unsigned int r = u + 0x7FFFu + ((u >> 16) & 1u);   // round-to-nearest-even
  return (unsigned short)(r >> 16);
}

// gfx1250 async global->LDS copy (ASYNCcnt-tracked), GV addressing mode.
// lds_off = wave-relative LDS byte address (low 32 bits of flat LDS pointer).
__device__ __forceinline__ void async_ld_b128(const unsigned short* g, unsigned lds_off) {
  asm volatile("global_load_async_to_lds_b128 %0, %1, off"
               :: "v"(lds_off), "v"(g)
               : "memory");
}
__device__ __forceinline__ unsigned lds_addr32(const void* p) {
  return (unsigned)(unsigned long long)p;   // LDS aperture: offset lives in addr[31:0]
}
__device__ __forceinline__ void wait_async_le3() {
  asm volatile("s_wait_asynccnt 3" ::: "memory");
}
__device__ __forceinline__ void wait_async_0() {
  asm volatile("s_wait_asynccnt 0" ::: "memory");
}

#define BB 4
#define TT 2048
#define DM 1024
#define DA 128

// ---------------- f32 -> bf16 conversion ----------------
__global__ void cvt_f32_bf16(const float* __restrict__ in,
                             unsigned short* __restrict__ out, int n) {
  int i = blockIdx.x * blockDim.x + threadIdx.x;
  int stride = gridDim.x * blockDim.x;
  for (; i < n; i += stride) out[i] = f32_to_bf16(in[i]);
}

// ---------------- bf16 GEMM: C = alpha * A[M,K] * Bt[N,K]^T ----------------
// block = 256 threads = 8 waves arranged 2(M) x 4(N); wave tile 32x32;
// block tile 64(M) x 128(N). A/B tiles double-buffered in LDS via
// global_load_async_to_lds_b128.  vmode==1 stores C transposed per batch:
// C[b][n][t] with t = m % 2048 (for Vt layout).
__global__ __launch_bounds__(256) void gemm_bf16(
    const unsigned short* __restrict__ A,
    const unsigned short* __restrict__ Bt,
    unsigned short* __restrict__ C,
    int M, int N, int K, float alpha, int vmode)
{
  __shared__ __align__(16) unsigned short As[2][64 * 32];    // 2 x 4 KB
  __shared__ __align__(16) unsigned short Bs[2][128 * 32];   // 2 x 8 KB

  const int tid  = threadIdx.x;
  const int lane = tid & 31;
  const int wave = tid >> 5;
  const int wm = wave >> 2;               // 0..1
  const int wn = wave & 3;                // 0..3
  const int mBase = blockIdx.y * 64;
  const int nBase = blockIdx.x * 128;
  const int lm = lane & 15;
  const int lh = lane >> 4;               // lane half (0/1)
  const long ldk = K;

  // Issue one K-step of async tile copies into buffer `buf`.
  // 3 async instructions per wave per stage (1 for A, 2 for B).
  auto stage = [&](int buf, int k0) {
    {
      int rowA = tid >> 2;                // 0..63
      int c    = tid & 3;                 // 16B chunk within 64B row
      const unsigned short* gp = A + (long)(mBase + rowA) * ldk + k0 + c * 8;
      async_ld_b128(gp, lds_addr32(&As[buf][rowA * 32 + c * 8]));
    }
#pragma unroll
    for (int u = 0; u < 2; u++) {
      int cid  = tid * 2 + u;             // 0..511
      int rowB = cid >> 2;                // 0..127
      int c    = cid & 3;
      const unsigned short* gp = Bt + (long)(nBase + rowB) * ldk + k0 + c * 8;
      async_ld_b128(gp, lds_addr32(&Bs[buf][rowB * 32 + c * 8]));
    }
  };

  const f32x8 zero = {0.f,0.f,0.f,0.f,0.f,0.f,0.f,0.f};
  f32x8 acc[2][2];
  acc[0][0] = zero; acc[0][1] = zero; acc[1][0] = zero; acc[1][1] = zero;

  stage(0, 0);
  int buf = 0;
  for (int k0 = 0; k0 < K; k0 += 32) {
    if (k0 + 32 < K) {
      stage(buf ^ 1, k0 + 32);            // prefetch next tile (async engine)
      if (k0 + 64 < K) {                  // warm L2 two tiles ahead
        __builtin_prefetch(A  + (long)(mBase + lm) * ldk + k0 + 64, 0, 3);
        __builtin_prefetch(Bt + (long)(nBase + lm) * ldk + k0 + 64, 0, 3);
      }
      wait_async_le3();                   // current stage landed (in-order)
    } else {
      wait_async_0();
    }
    __syncthreads();

    FragB16 a[2], b[2];
#pragma unroll
    for (int i = 0; i < 2; i++) {
      const unsigned short* ap = &As[buf][(wm*32 + i*16 + lm) * 32 + lh*8];
      a[i].q[0] = *(const uint4*)(ap);
      a[i].q[1] = *(const uint4*)(ap + 16);
    }
#pragma unroll
    for (int j = 0; j < 2; j++) {
      const unsigned short* bp = &Bs[buf][(wn*32 + j*16 + lm) * 32 + lh*8];
      b[j].q[0] = *(const uint4*)(bp);
      b[j].q[1] = *(const uint4*)(bp + 16);
    }
#pragma unroll
    for (int i = 0; i < 2; i++)
#pragma unroll
      for (int j = 0; j < 2; j++)
        acc[i][j] = __builtin_amdgcn_wmma_f32_16x16x32_bf16(
            false, a[i].v, false, b[j].v, (short)0, acc[i][j], false, false);

    __syncthreads();                      // done reading `buf` before re-fill
    buf ^= 1;
  }

#pragma unroll
  for (int i = 0; i < 2; i++)
#pragma unroll
    for (int j = 0; j < 2; j++)
#pragma unroll
      for (int r = 0; r < 8; r++) {
        int m = mBase + wm*32 + i*16 + r + lh*8;
        int n = nBase + wn*32 + j*16 + lm;
        unsigned short hv = f32_to_bf16(alpha * acc[i][j][r]);
        if (vmode == 0) {
          C[(long)m * N + n] = hv;
        } else {
          int bb = m >> 11;               // T = 2048
          int tm = m & 2047;
          C[((long)bb * N + n) * TT + tm] = hv;
        }
      }
}

// ---------------- Flash attention (causal, online softmax) ----------------
// grid: (DM/128, T/128, B); block 256 = 8 waves.
// Wave w: 16 query rows (qbase + 16*w), 128-wide output column tile.
__global__ __launch_bounds__(256) void flash_attn(
    const unsigned short* __restrict__ Qb,   // [B*T, 128] bf16, pre-scaled 1/sqrt(128)
    const unsigned short* __restrict__ Kb,   // [B*T, 128] bf16
    const unsigned short* __restrict__ Vt,   // [B, 1024, T] bf16 (V transposed)
    float* __restrict__ out)                 // [B, T, 1024] f32
{
  __shared__ __align__(16) unsigned short Plds[8][16 * 32]; // per-wave P tile
  const int lane = threadIdx.x & 31;
  const int wave = threadIdx.x >> 5;
  const int lm = lane & 15;
  const int lh = lane >> 4;
  const int b  = blockIdx.z;
  const int qw = blockIdx.y * 128 + wave * 16;  // wave's query-row base
  const int cb = blockIdx.x * 128;              // output column base

  // Q fragments: 16 rows x 128 (4 A-fragments of 16x32)
  FragB16 qf[4];
  {
    const unsigned short* qp = Qb + (long)(b*TT + qw + lm) * DA + lh*8;
#pragma unroll
    for (int kk = 0; kk < 4; kk++) {
      qf[kk].q[0] = *(const uint4*)(qp + kk*32);
      qf[kk].q[1] = *(const uint4*)(qp + kk*32 + 16);
    }
  }

  const f32x8 zero = {0.f,0.f,0.f,0.f,0.f,0.f,0.f,0.f};
  f32x8 accO[8];
#pragma unroll
  for (int j = 0; j < 8; j++) accO[j] = zero;
  float mrun[8], lrun[8];
#pragma unroll
  for (int r = 0; r < 8; r++) { mrun[r] = -3.0e38f; lrun[r] = 0.0f; }

  const unsigned short* vbase = Vt + (long)b * DM * TT;
  const int send = qw + 16;                    // causal bound for this wave
  for (int s = 0; s < send; s += 32) {
    // ---- scores: S[16 x 32] = Q (16x128) . K^T, two 16x16 C fragments ----
    f32x8 sc[2];
    sc[0] = zero; sc[1] = zero;
#pragma unroll
    for (int j = 0; j < 2; j++) {
      FragB16 kf[4];
      const unsigned short* kp = Kb + (long)(b*TT + s + j*16 + lm) * DA + lh*8;
#pragma unroll
      for (int kk = 0; kk < 4; kk++) {
        kf[kk].q[0] = *(const uint4*)(kp + kk*32);
        kf[kk].q[1] = *(const uint4*)(kp + kk*32 + 16);
      }
#pragma unroll
      for (int kk = 0; kk < 4; kk++)
        sc[j] = __builtin_amdgcn_wmma_f32_16x16x32_bf16(
            false, qf[kk].v, false, kf[kk].v, (short)0, sc[j], false, false);
    }

    // ---- causal mask + online softmax; stage P into LDS in A-layout order ----
#pragma unroll
    for (int r = 0; r < 8; r++) {
      int qrow = qw + r + lh*8;
      float s0 = ((s + lm)      <= qrow) ? sc[0][r] : -3.0e38f;
      float s1 = ((s + 16 + lm) <= qrow) ? sc[1][r] : -3.0e38f;
      float tmax = fmaxf(s0, s1);
#pragma unroll
      for (int off = 8; off >= 1; off >>= 1)
        tmax = fmaxf(tmax, __shfl_xor(tmax, off, 32));
      float nm = fmaxf(mrun[r], tmax);
      float p0 = __expf(s0 - nm);
      float p1 = __expf(s1 - nm);
      float corr = __expf(mrun[r] - nm);
      mrun[r] = nm;
      float ps = p0 + p1;
#pragma unroll
      for (int off = 8; off >= 1; off >>= 1)
        ps += __shfl_xor(ps, off, 32);
      lrun[r] = lrun[r] * corr + ps;
#pragma unroll
      for (int j = 0; j < 8; j++) accO[j][r] *= corr;
      Plds[wave][(r + lh*8)*32 + lm]      = f32_to_bf16(p0);
      Plds[wave][(r + lh*8)*32 + 16 + lm] = f32_to_bf16(p1);
    }

    // ---- reload P (16x32) as WMMA A fragment (DS ops in-order within wave) ----
    FragB16 pf;
    pf.q[0] = *(const uint4*)(&Plds[wave][lm*32 + lh*8]);
    pf.q[1] = *(const uint4*)(&Plds[wave][lm*32 + 16 + lh*8]);

    // ---- accO += P (16x32) . V[s:s+32, cb:cb+128] ----
#pragma unroll
    for (int j = 0; j < 8; j++) {
      FragB16 vf;
      const unsigned short* vp = vbase + (long)(cb + j*16 + lm) * TT + s + lh*8;
      vf.q[0] = *(const uint4*)(vp);
      vf.q[1] = *(const uint4*)(vp + 16);
      accO[j] = __builtin_amdgcn_wmma_f32_16x16x32_bf16(
          false, pf.v, false, vf.v, (short)0, accO[j], false, false);
    }
  }

  // ---- epilogue: normalize and store f32 output ----
#pragma unroll
  for (int r = 0; r < 8; r++) {
    float inv = 1.0f / lrun[r];
    int qrow = qw + r + lh*8;
    float* op = out + ((long)b*TT + qrow) * DM + cb + lm;
#pragma unroll
    for (int j = 0; j < 8; j++) op[j*16] = accO[j][r] * inv;
  }
}

// ---------------- Host launch ----------------
extern "C" void kernel_launch(void* const* d_in, const int* in_sizes, int n_in,
                              void* d_out, int out_size, void* d_ws, size_t ws_size,
                              hipStream_t stream) {
  const float* x  = (const float*)d_in[0];
  const float* Wq = (const float*)d_in[1];
  const float* Wk = (const float*)d_in[2];
  const float* Wv = (const float*)d_in[3];
  float* out = (float*)d_out;

  char* ws = (char*)d_ws;
  // Workspace layout (bytes):
  unsigned short* xb  = (unsigned short*)(ws);                     // 16 MB
  unsigned short* Wqb = (unsigned short*)(ws + 16777216);          // 256 KB
  unsigned short* Wkb = (unsigned short*)(ws + 17039360);          // 256 KB
  unsigned short* Wvb = (unsigned short*)(ws + 17301504);          // 2 MB
  unsigned short* Qb  = (unsigned short*)(ws + 19398656);          // 2 MB
  unsigned short* Kb  = (unsigned short*)(ws + 21495808);          // 2 MB
  unsigned short* Vt  = (unsigned short*)(ws + 23592960);          // 16 MB

  auto cvt = [&](const float* in, unsigned short* o, int n) {
    int blocks = (n + 255) / 256;
    if (blocks > 8192) blocks = 8192;
    cvt_f32_bf16<<<blocks, 256, 0, stream>>>(in, o, n);
  };
  cvt(x,  xb,  BB * TT * DM);   // 8,388,608
  cvt(Wq, Wqb, DA * DM);        // 131,072
  cvt(Wk, Wkb, DA * DM);
  cvt(Wv, Wvb, DM * DM);        // 1,048,576

  const int M = BB * TT;        // 8192
  const float qscale = 0.08838834764831845f;  // 1/sqrt(128)

  // Q = (x Wq^T) * qscale ; K = x Wk^T ; V = x Wv^T (stored transposed per batch)
  gemm_bf16<<<dim3(1, M / 64), 256, 0, stream>>>(xb, Wqb, Qb, M, DA, DM, qscale, 0);
  gemm_bf16<<<dim3(1, M / 64), 256, 0, stream>>>(xb, Wkb, Kb, M, DA, DM, 1.0f, 0);
  gemm_bf16<<<dim3(DM / 128, M / 64), 256, 0, stream>>>(xb, Wvb, Vt, M, DM, DM, 1.0f, 1);

  // Causal flash attention: grid (col tiles, q tiles, batch)
  flash_attn<<<dim3(DM / 128, TT / 128, BB), 256, 0, stream>>>(Qb, Kb, Vt, out);
}
